// DINAttenLayer_19172734009920
// MI455X (gfx1250) — matrix operand order repaired and checked by reading, not compiled, not run on hardware
//
#include <hip/hip_runtime.h>
#include <hip/hip_bf16.h>

typedef __attribute__((ext_vector_type(16))) __bf16 v16bf;
typedef __attribute__((ext_vector_type(8)))  __bf16 v8bf;
typedef __attribute__((ext_vector_type(8)))  float  v8f;

#define B_   2048
#define T_   200
#define E_   64
#define H1   80
#define H2   40
#define NT1  5    // 80 / 16 N-tiles for GEMM1
#define KT1  8    // 256 / 32 K-tiles for GEMM1
#define NT2  3    // 48 / 16 N-tiles for GEMM2 (40 padded to 48)
#define KT2  3    // 96 / 32 K-tiles for GEMM2 (80 padded to 96)
#define CHUNKS 13 // ceil(200/16)
#define THREADS (CHUNKS * 32)
#define PADV (-4294967295.0f)

#define W1ELEMS (NT1 * KT1 * 512)   // 20480 bf16 = 40 KB
#define W2ELEMS (NT2 * KT2 * 512)   //  4608 bf16 =  9 KB

__device__ __forceinline__ float sigmoidf_(float x) {
    return 1.0f / (1.0f + __expf(-x));
}

// ---------------------------------------------------------------------------
// Repack W1 (256x80 f32, row-major K x N) into WMMA-B bf16 fragment order:
// fragment fi = n*KT1 + j; within fragment: lane L, element e:
//   K = j*32 + (L/16)*8 + (e%8) + (e/8)*16,  N = n*16 + (L%16)
// so each lane's 16 elements are two 8-element K runs.
// ---------------------------------------------------------------------------
__global__ void pack_w1_kernel(const float* __restrict__ W1,
                               unsigned short* __restrict__ w1p) {
    int idx = blockIdx.x * 256 + threadIdx.x;
    if (idx >= W1ELEMS) return;
    int fi = idx >> 9;
    int L  = (idx >> 4) & 31;
    int e  = idx & 15;
    int n = fi / KT1, j = fi % KT1;
    int K = j * 32 + (L >> 4) * 8 + (e & 7) + (e >> 3) * 16;
    int N = n * 16 + (L & 15);
    __bf16 h = (__bf16)W1[K * H1 + N];
    w1p[idx] = __builtin_bit_cast(unsigned short, h);
}

// W2 (80x40 f32) -> padded 96x48 bf16 fragments, zeros outside.
__global__ void pack_w2_kernel(const float* __restrict__ W2,
                               unsigned short* __restrict__ w2p) {
    int idx = blockIdx.x * 256 + threadIdx.x;
    if (idx >= W2ELEMS) return;
    int fi = idx >> 9;
    int L  = (idx >> 4) & 31;
    int e  = idx & 15;
    int n = fi / KT2, k = fi % KT2;
    int K = k * 32 + (L >> 4) * 8 + (e & 7) + (e >> 3) * 16;
    int N = n * 16 + (L & 15);
    float v = (K < H1 && N < H2) ? W2[K * H2 + N] : 0.0f;
    __bf16 h = (__bf16)v;
    w2p[idx] = __builtin_bit_cast(unsigned short, h);
}

// ---------------------------------------------------------------------------
// Main kernel: one workgroup per batch element b, 13 waves, wave w owns
// T-chunk rows [w*16, w*16+16).
// ---------------------------------------------------------------------------
__global__ __launch_bounds__(THREADS) void din_attn_kernel(
    const float* __restrict__ q, const float* __restrict__ facts,
    const int* __restrict__ mask,
    const unsigned short* __restrict__ w1p_u, const float* __restrict__ b1,
    const unsigned short* __restrict__ w2p_u, const float* __restrict__ b2,
    const float* __restrict__ W3, const float* __restrict__ b3,
    float* __restrict__ out)
{
    __shared__ __align__(16) __bf16 w1s[W1ELEMS];        // LDS copy of W1 frags
    __shared__ __align__(16) __bf16 w2s[W2ELEMS];        // LDS copy of W2 frags
    __shared__ __align__(16) __bf16 hl[CHUNKS][16 * 96]; // per-wave h1 tile
    __shared__ float scores[CHUNKS * 16];                // 208 scores
    __shared__ float part[CHUNKS][E_];                   // weighted-sum partials
    __shared__ float redMax, redSum;

    const int tid = threadIdx.x;
    const int wv  = tid >> 5;   // wave = T-chunk index (0..12)
    const int L   = tid & 31;   // lane
    const int g   = L >> 4;     // lane group (K-half select)
    const int lid = L & 15;     // lane-in-group (N / M select)
    const int b   = blockIdx.x;

    // ---- cooperative copy of packed weights into LDS (issued first) --------
    {
        const uint4* src1 = (const uint4*)w1p_u;
        uint4*       dst1 = (uint4*)w1s;
        for (int i = tid; i < W1ELEMS / 8; i += THREADS) dst1[i] = src1[i];
        const uint4* src2 = (const uint4*)w2p_u;
        uint4*       dst2 = (uint4*)w2s;
        for (int i = tid; i < W2ELEMS / 8; i += THREADS) dst2[i] = src2[i];
    }

    // ---- hoisted small-constant loads (issue early, all L2-resident) -------
    float bias1[NT1];
    #pragma unroll
    for (int n = 0; n < NT1; ++n) bias1[n] = b1[n * 16 + lid];
    float bias2[NT2], w3v[NT2];
    #pragma unroll
    for (int n = 0; n < NT2; ++n) {
        int col = n * 16 + lid;
        bias2[n] = (col < H2) ? b2[col] : 0.0f;
        w3v[n]   = (col < H2) ? W3[col] : 0.0f;
    }
    float b3v = b3[0];

    // ---- load q / facts runs for this lane's A-matrix row ------------------
    // lane L feeds row trow = wv*16 + lid; group g covers K offsets
    // {g*8 + 16m : m=0..3} within each 64-wide din block.
    const int trow = wv * 16 + lid;
    const float* qb = q + (size_t)b * E_;
    const float* fb = facts + ((size_t)b * T_ + trow) * E_;
    float qs[4][8], fs[4][8];
    #pragma unroll
    for (int m = 0; m < 4; ++m) {
        int o = g * 8 + m * 16;
        float4 a0 = *(const float4*)(qb + o);
        float4 a1 = *(const float4*)(qb + o + 4);
        qs[m][0] = a0.x; qs[m][1] = a0.y; qs[m][2] = a0.z; qs[m][3] = a0.w;
        qs[m][4] = a1.x; qs[m][5] = a1.y; qs[m][6] = a1.z; qs[m][7] = a1.w;
        if (trow < T_) {
            float4 c0 = *(const float4*)(fb + o);
            float4 c1 = *(const float4*)(fb + o + 4);
            fs[m][0] = c0.x; fs[m][1] = c0.y; fs[m][2] = c0.z; fs[m][3] = c0.w;
            fs[m][4] = c1.x; fs[m][5] = c1.y; fs[m][6] = c1.z; fs[m][7] = c1.w;
        } else {
            #pragma unroll
            for (int e = 0; e < 8; ++e) fs[m][e] = 0.0f;
        }
    }

    // ---- build all 8 A fragments of din = [q | f | q-f | q*f] in registers -
    // K-tile j: block s=j>>1 (0:q 1:f 2:q-f 3:q*f); runs at slots m1=2*(j&1), m1+1.
    v16bf afr[KT1];
    #pragma unroll
    for (int j = 0; j < KT1; ++j) {
        const int s  = j >> 1;
        const int m1 = (j & 1) * 2;
        const int m2 = m1 + 1;
        #pragma unroll
        for (int e = 0; e < 8; ++e) {
            float x1, x2;
            if (s == 0)      { x1 = qs[m1][e];              x2 = qs[m2][e]; }
            else if (s == 1) { x1 = fs[m1][e];              x2 = fs[m2][e]; }
            else if (s == 2) { x1 = qs[m1][e] - fs[m1][e];  x2 = qs[m2][e] - fs[m2][e]; }
            else             { x1 = qs[m1][e] * fs[m1][e];  x2 = qs[m2][e] * fs[m2][e]; }
            afr[j][e]     = (__bf16)x1;
            afr[j][e + 8] = (__bf16)x2;
        }
    }

    __syncthreads();   // weights resident in LDS from here on

    // ---- GEMM1: (16x256) @ (256x80) + b1, sigmoid, stage bf16 tile in LDS --
    // B fragments now come from LDS: 2x ds_load_b128 per WMMA, short latency.
    #pragma unroll
    for (int n = 0; n < NT1; ++n) {
        v8f c;
        #pragma unroll
        for (int r = 0; r < 8; ++r) c[r] = bias1[n];
        #pragma unroll
        for (int j = 0; j < KT1; ++j) {
            v16bf bfrag = *(const v16bf*)(w1s + ((n * KT1 + j) * 512 + L * 16));
            c = __builtin_amdgcn_wmma_f32_16x16x32_bf16(
                    false, afr[j], false, bfrag, (short)0, c, false, false);
        }
        #pragma unroll
        for (int r = 0; r < 8; ++r) {
            float h = sigmoidf_(c[r]);
            hl[wv][(r + 8 * g) * 96 + n * 16 + lid] = (__bf16)h;
        }
    }
    // zero-pad h1 columns 80..95 (wave-private tile; same-wave DS ops are in order)
    #pragma unroll
    for (int r = 0; r < 8; ++r)
        hl[wv][(r + 8 * g) * 96 + 80 + lid] = (__bf16)0.0f;

    // ---- GEMM2: (16x96) @ (96x48) + b2 ------------------------------------
    v8f d2[NT2];
    #pragma unroll
    for (int n = 0; n < NT2; ++n) {
        v8f c;
        #pragma unroll
        for (int r = 0; r < 8; ++r) c[r] = bias2[n];
        #pragma unroll
        for (int k = 0; k < KT2; ++k) {
            v8bf r1 = *(const v8bf*)(&hl[wv][lid * 96 + k * 32 + g * 8]);
            v8bf r2 = *(const v8bf*)(&hl[wv][lid * 96 + k * 32 + g * 8 + 16]);
            v16bf a2;
            #pragma unroll
            for (int e = 0; e < 8; ++e) { a2[e] = r1[e]; a2[e + 8] = r2[e]; }
            v16bf bfrag = *(const v16bf*)(w2s + ((n * KT2 + k) * 512 + L * 16));
            c = __builtin_amdgcn_wmma_f32_16x16x32_bf16(
                    false, a2, false, bfrag, (short)0, c, false, false);
        }
        d2[n] = c;
    }

    // ---- scores: sigmoid(h2) @ W3 + b3, cross-lane reduce, mask ------------
    float sc[8];
    #pragma unroll
    for (int r = 0; r < 8; ++r)
        sc[r] = sigmoidf_(d2[0][r]) * w3v[0]
              + sigmoidf_(d2[1][r]) * w3v[1]
              + sigmoidf_(d2[2][r]) * w3v[2];
    #pragma unroll
    for (int m = 1; m < 16; m <<= 1) {
        #pragma unroll
        for (int r = 0; r < 8; ++r)
            sc[r] += __shfl_xor(sc[r], m, 32);
    }
    if (lid == 0) { // lanes 0 and 16 hold rows 8g..8g+7
        #pragma unroll
        for (int r = 0; r < 8; ++r) {
            int t = wv * 16 + 8 * g + r;
            bool ok = (t < T_) && (mask[(size_t)b * T_ + t] == 1);
            scores[t] = ok ? (sc[r] + b3v) : PADV;
        }
    }
    __syncthreads();

    // ---- softmax over 208 scores ------------------------------------------
    if (wv == 0) {
        float m = -3.4e38f;
        for (int t = L; t < CHUNKS * 16; t += 32) m = fmaxf(m, scores[t]);
        #pragma unroll
        for (int o = 16; o >= 1; o >>= 1) m = fmaxf(m, __shfl_xor(m, o, 32));
        if (L == 0) redMax = m;
    }
    __syncthreads();
    float mx = redMax;
    if (tid < CHUNKS * 16) scores[tid] = __expf(scores[tid] - mx);
    __syncthreads();
    if (wv == 0) {
        float s = 0.0f;
        for (int t = L; t < CHUNKS * 16; t += 32) s += scores[t];
        #pragma unroll
        for (int o = 16; o >= 1; o >>= 1) s += __shfl_xor(s, o, 32);
        if (L == 0) redSum = s;
    }

    // ---- weighted sum of facts: each wave handles its 16 rows --------------
    float ax = 0.0f, ay = 0.0f;
    for (int r = 0; r < 16; ++r) {
        int t = wv * 16 + r;
        if (t < T_) {
            float a = scores[t];
            float2 f2 = *(const float2*)(facts + ((size_t)b * T_ + t) * E_ + 2 * L);
            ax += a * f2.x;
            ay += a * f2.y;
        }
    }
    part[wv][2 * L]     = ax;
    part[wv][2 * L + 1] = ay;
    __syncthreads();

    if (tid < E_) {
        float s = 0.0f;
        #pragma unroll
        for (int w = 0; w < CHUNKS; ++w) s += part[w][tid];
        out[(size_t)b * E_ + tid] = s / redSum;
    }
}

extern "C" void kernel_launch(void* const* d_in, const int* in_sizes, int n_in,
                              void* d_out, int out_size, void* d_ws, size_t ws_size,
                              hipStream_t stream) {
    const float* q     = (const float*)d_in[0];
    const float* facts = (const float*)d_in[1];
    const int*   mask  = (const int*)  d_in[2];
    const float* W1    = (const float*)d_in[3];
    const float* b1    = (const float*)d_in[4];
    const float* W2    = (const float*)d_in[5];
    const float* b2    = (const float*)d_in[6];
    const float* W3    = (const float*)d_in[7];
    const float* b3    = (const float*)d_in[8];
    float* out = (float*)d_out;

    unsigned short* w1p = (unsigned short*)d_ws;                       // 40 KB
    unsigned short* w2p = (unsigned short*)((char*)d_ws + 64 * 1024);  //  9 KB

    pack_w1_kernel<<<(W1ELEMS + 255) / 256, 256, 0, stream>>>(W1, w1p);
    pack_w2_kernel<<<(W2ELEMS + 255) / 256, 256, 0, stream>>>(W2, w2p);
    din_attn_kernel<<<B_, THREADS, 0, stream>>>(q, facts, mask,
                                                w1p, b1, w2p, b2, W3, b3, out);
}
